// MicroConv_74835510165572
// MI455X (gfx1250) — matrix operand order repaired and checked by reading, not compiled, not run on hardware
//
#include <hip/hip_runtime.h>
#include <hip/hip_bf16.h>

#define N_SRC    50000
#define N_DST    50000
#define N_EDGES  800000
#define IN_DIM   128
#define NUM_HEADS 8
#define OUT_DIM  16
#define HD       128          // NUM_HEADS*OUT_DIM
#define NEG_SLOPE 0.2f

typedef _Float16 v16h __attribute__((ext_vector_type(16)));
typedef float    v8f  __attribute__((ext_vector_type(8)));

__device__ __forceinline__ float lrelu(float x) { return x > 0.f ? x : NEG_SLOPE * x; }

__device__ __forceinline__ void cvt8(v16h& v, int base, float4 a, float4 b) {
    v[base+0]=(_Float16)a.x; v[base+1]=(_Float16)a.y; v[base+2]=(_Float16)a.z; v[base+3]=(_Float16)a.w;
    v[base+4]=(_Float16)b.x; v[base+5]=(_Float16)b.y; v[base+6]=(_Float16)b.z; v[base+7]=(_Float16)b.w;
}

// ---------------------------------------------------------------------------
// K0: one-shot weight prep.
//   * W (f32) -> Wh (f16), per side (removes all B-side converts from GEMM).
//   * Fold attention vector into the projection:
//       Wcomb[side][h][k] = sum_d attn[h, side*16+d] * W_side[h*16+d][k]
//       bcomb[side][h]    = sum_d attn[h, side*16+d] * b_side[h*16+d]
//     padded to 16 rows (rows 8..15 = 0) so the logits become one extra WMMA
//     N-tile of the main GEMM -- no cross-lane reductions needed.
// ---------------------------------------------------------------------------
__global__ __launch_bounds__(256)
void setup_weights(const float* __restrict__ Ws, const float* __restrict__ Wd,
                   const float* __restrict__ bs, const float* __restrict__ bd,
                   const float* __restrict__ attn,
                   _Float16* __restrict__ Whs, _Float16* __restrict__ Whd,
                   _Float16* __restrict__ Wcomb,   // [2][16][128] f16, padded
                   float* __restrict__ bcomb)      // [2][16], padded
{
    const int i = blockIdx.x * blockDim.x + threadIdx.x;
    if (i < IN_DIM * HD) {
        Whs[i] = (_Float16)Ws[i];
        Whd[i] = (_Float16)Wd[i];
    }
    if (i < 2 * 16 * IN_DIM) {
        const int side = i >> 11;          // / (16*128)
        const int rem  = i & 2047;
        const int n    = rem >> 7;         // padded row 0..15 (head if < 8)
        const int k    = rem & 127;
        const float* W = side ? Wd : Ws;
        float v = 0.f;
        if (n < NUM_HEADS) {
            #pragma unroll
            for (int d = 0; d < OUT_DIM; ++d)
                v += attn[n*32 + side*16 + d] * W[(size_t)(n*16 + d) * IN_DIM + k];
        }
        Wcomb[i] = (_Float16)v;
        if (k == 0) {
            const float* b = side ? bd : bs;
            float bb = 0.f;
            if (n < NUM_HEADS) {
                #pragma unroll
                for (int d = 0; d < OUT_DIM; ++d)
                    bb += attn[n*32 + side*16 + d] * b[n*16 + d];
            }
            bcomb[side*16 + n] = bb;
        }
    }
}

// ---------------------------------------------------------------------------
// K1: fused projection GEMM (WMMA f16->f32) + attention logits as a 9th N-tile.
// grid.y==0: h_src stored + el; grid.y==1: er only (h_dst never stored).
// One wave computes a 16x128 row-tile: 8 head tiles + 1 logit tile, 4 WMMAs each.
// ---------------------------------------------------------------------------
__global__ __launch_bounds__(256)
void gat_proj_kernel(const float* __restrict__ feat_src, const float* __restrict__ feat_dst,
                     const _Float16* __restrict__ Wh_src, const float* __restrict__ b_src,
                     const _Float16* __restrict__ Wh_dst, const float* __restrict__ b_dst,
                     const _Float16* __restrict__ Wcomb,  const float* __restrict__ bcomb,
                     float* __restrict__ h_src, float* __restrict__ el, float* __restrict__ er)
{
    const int side = blockIdx.y;                       // 0 = src, 1 = dst
    const float*    feat = side ? feat_dst : feat_src;
    const _Float16* Wh   = side ? Wh_dst   : Wh_src;
    const float*    bias = side ? b_dst    : b_src;
    const _Float16* WcS  = Wcomb + (size_t)side * 16 * IN_DIM;
    const float*    bcS  = bcomb + side * 16;
    float* logit         = side ? er       : el;

    const int wave = threadIdx.x >> 5;
    const int lane = threadIdx.x & 31;
    const int tile = blockIdx.x * 8 + wave;
    if (tile >= N_SRC / 16) return;                    // uniform per-wave (EXEC stays full)
    const int R  = tile * 16;
    const int n  = lane & 15;                          // A-row / B-col / D-col index
    const int kh = lane >> 4;                          // K-half selector

    // --- A tile: 16 rows x 128 K, four 16x32 f16 chunks in WMMA A layout ---
    v16h a[4];
    const float* rowA = feat + (size_t)(R + n) * IN_DIM;
    #pragma unroll
    for (int kc4 = 0; kc4 < 4; ++kc4) {
        const int kc = kc4 * 32;
        float4 p0 = *(const float4*)(rowA + kc + kh*16);
        float4 p1 = *(const float4*)(rowA + kc + kh*16 + 4);
        float4 p2 = *(const float4*)(rowA + kc + kh*16 + 8);
        float4 p3 = *(const float4*)(rowA + kc + kh*16 + 12);
        cvt8(a[kc4], 0, p0, p1);
        cvt8(a[kc4], 8, p2, p3);
    }

    // --- 8 head tiles: D = A x Wh^T + bias ---
    #pragma unroll 1
    for (int h = 0; h < NUM_HEADS; ++h) {
        v8f c;
        const float bv = bias[h*16 + n];
        #pragma unroll
        for (int r = 0; r < 8; ++r) c[r] = bv;

        const _Float16* rowB = Wh + (size_t)(h*16 + n) * IN_DIM;
        #pragma unroll
        for (int kc4 = 0; kc4 < 4; ++kc4) {
            const v16h b = *(const v16h*)(rowB + kc4*32 + kh*16);
            c = __builtin_amdgcn_wmma_f32_16x16x32_f16(false, a[kc4], false, b,
                                                       (short)0, c, false, false);
        }
        if (side == 0) {
            #pragma unroll
            for (int r = 0; r < 8; ++r)
                h_src[(size_t)(R + kh*8 + r) * HD + h*16 + n] = c[r];
        }
    }

    // --- logit tile: D columns 0..7 = el/er[row][head] directly ---
    {
        v8f c;
        const float bv = bcS[n];                       // 0 for padded cols n>=8
        #pragma unroll
        for (int r = 0; r < 8; ++r) c[r] = bv;

        const _Float16* rowB = WcS + (size_t)n * IN_DIM;
        #pragma unroll
        for (int kc4 = 0; kc4 < 4; ++kc4) {
            const v16h b = *(const v16h*)(rowB + kc4*32 + kh*16);
            c = __builtin_amdgcn_wmma_f32_16x16x32_f16(false, a[kc4], false, b,
                                                       (short)0, c, false, false);
        }
        if (n < NUM_HEADS) {
            #pragma unroll
            for (int r = 0; r < 8; ++r)
                logit[(size_t)(R + kh*8 + r) * NUM_HEADS + n] = c[r];
        }
    }
}

// ---------------------------------------------------------------------------
// CSR-by-destination build: count -> scan -> fill
// ---------------------------------------------------------------------------
__global__ void zero_i32(int* __restrict__ p, int count) {
    int i = blockIdx.x * blockDim.x + threadIdx.x;
    if (i < count) p[i] = 0;
}

__global__ void edge_count(const int* __restrict__ dst_idx, int* __restrict__ counts) {
    int e = blockIdx.x * blockDim.x + threadIdx.x;
    if (e < N_EDGES) atomicAdd(&counts[dst_idx[e]], 1);
}

#define SCAN_T 512
__global__ __launch_bounds__(SCAN_T)
void scan1(const int* __restrict__ counts, int* __restrict__ offs, int* __restrict__ bsums) {
    __shared__ int sm[SCAN_T];
    const int t = threadIdx.x, g = blockIdx.x * SCAN_T + t;
    const int x = (g < N_DST) ? counts[g] : 0;
    sm[t] = x; __syncthreads();
    for (int off = 1; off < SCAN_T; off <<= 1) {
        int v = (t >= off) ? sm[t - off] : 0;
        __syncthreads();
        sm[t] += v;
        __syncthreads();
    }
    if (g < N_DST) offs[g] = sm[t] - x;               // block-local exclusive
    if (t == SCAN_T - 1) bsums[blockIdx.x] = sm[t];
}

__global__ __launch_bounds__(128)
void scan2(int* __restrict__ bsums, int nb) {         // nb <= 128
    __shared__ int sm[128];
    const int t = threadIdx.x;
    const int x = (t < nb) ? bsums[t] : 0;
    sm[t] = x; __syncthreads();
    for (int off = 1; off < 128; off <<= 1) {
        int v = (t >= off) ? sm[t - off] : 0;
        __syncthreads();
        sm[t] += v;
        __syncthreads();
    }
    if (t < nb) bsums[t] = sm[t] - x;                 // exclusive block offsets
}

__global__ __launch_bounds__(SCAN_T)
void scan3(int* __restrict__ offs, const int* __restrict__ bsums, int* __restrict__ cursor) {
    const int t = threadIdx.x, g = blockIdx.x * SCAN_T + t;
    if (g < N_DST) {
        const int o = offs[g] + bsums[blockIdx.x];
        offs[g]   = o;
        cursor[g] = o;
    }
    if (g == 0) offs[N_DST] = N_EDGES;
}

__global__ void edge_fill(const int* __restrict__ src_idx, const int* __restrict__ dst_idx,
                          int* __restrict__ cursor, int* __restrict__ csr_src) {
    int e = blockIdx.x * blockDim.x + threadIdx.x;
    if (e < N_EDGES) {
        int p = atomicAdd(&cursor[dst_idx[e]], 1);
        csr_src[p] = src_idx[e];
    }
}

// ---------------------------------------------------------------------------
// K6: one wave per destination node: segment max -> exp-sum + weighted feature
// accumulation in registers -> one 512B store. No float atomics, L2-resident.
// ---------------------------------------------------------------------------
__global__ __launch_bounds__(256)
void gat_agg(const float* __restrict__ h_src, const float* __restrict__ el,
             const float* __restrict__ er,    const int* __restrict__ offs,
             const int* __restrict__ csr_src, float* __restrict__ out)
{
    const int wave = threadIdx.x >> 5;
    const int lane = threadIdx.x & 31;
    const int d = blockIdx.x * 8 + wave;
    if (d >= N_DST) return;

    const int start = offs[d], end = offs[d + 1];

    // Phase A: per-head segment max, 4 edge slots x 8 heads across the wave
    const int e4 = lane >> 3;                   // edge slot 0..3
    const int hA = lane & 7;                    // head 0..7
    const float erA = er[(size_t)d * NUM_HEADS + hA];
    float mymax = -__builtin_inff();
    for (int i = start + e4; i < end; i += 4) {
        const int s = csr_src[i];
        mymax = fmaxf(mymax, lrelu(el[(size_t)s * NUM_HEADS + hA] + erA));
    }
    mymax = fmaxf(mymax, __shfl_xor(mymax, 8, 32));
    mymax = fmaxf(mymax, __shfl_xor(mymax, 16, 32));

    // Phase B layout: lane covers channels c0..c0+3, head hl = c0/16
    const int hl = lane >> 2;
    const int c0 = lane * 4;
    const float m_h  = __shfl(mymax, hl, 32);   // lane hl holds head-hl max
    const float er_d = er[(size_t)d * NUM_HEADS + hl];

    float expsum = 0.f;
    float acc0 = 0.f, acc1 = 0.f, acc2 = 0.f, acc3 = 0.f;
    for (int i = start; i < end; ++i) {
        const int s = csr_src[i];
        const float sc = lrelu(el[(size_t)s * NUM_HEADS + hl] + er_d);
        const float w  = expf(sc - m_h);
        expsum += w;
        const float4 hv = *(const float4*)(h_src + (size_t)s * HD + c0);
        acc0 += w * hv.x; acc1 += w * hv.y; acc2 += w * hv.z; acc3 += w * hv.w;
    }
    float denom = expsum;
    if (denom == 0.f) denom = 1.f;              // empty segments -> zeros
    const float inv = 1.f / denom;
    float4 o;
    o.x = acc0 * inv; o.y = acc1 * inv; o.z = acc2 * inv; o.w = acc3 * inv;
    *(float4*)(out + (size_t)d * HD + c0) = o;
}

// ---------------------------------------------------------------------------
extern "C" void kernel_launch(void* const* d_in, const int* in_sizes, int n_in,
                              void* d_out, int out_size, void* d_ws, size_t ws_size,
                              hipStream_t stream)
{
    const float* feat_src = (const float*)d_in[0];
    const float* feat_dst = (const float*)d_in[1];
    const float* W_src    = (const float*)d_in[2];
    const float* b_src    = (const float*)d_in[3];
    const float* W_dst    = (const float*)d_in[4];
    const float* b_dst    = (const float*)d_in[5];
    const float* attn     = (const float*)d_in[6];
    const int*   src_idx  = (const int*)d_in[7];
    const int*   dst_idx  = (const int*)d_in[8];
    float* out = (float*)d_out;
    (void)in_sizes; (void)n_in; (void)out_size; (void)ws_size;

    char* ws = (char*)d_ws;
    size_t off = 0;
    auto alloc = [&](size_t bytes) -> char* {
        char* p = ws + off;
        off += (bytes + 255) & ~(size_t)255;
        return p;
    };
    float*     h_src_buf = (float*)alloc((size_t)N_SRC * HD * sizeof(float));        // 25.6 MB
    float*     el_buf    = (float*)alloc((size_t)N_SRC * NUM_HEADS * sizeof(float)); // 1.6 MB
    float*     er_buf    = (float*)alloc((size_t)N_DST * NUM_HEADS * sizeof(float)); // 1.6 MB
    int*       counts    = (int*)alloc((size_t)N_DST * sizeof(int));  // reused as cursor
    int*       offs      = (int*)alloc((size_t)(N_DST + 1) * sizeof(int));
    int*       bsums     = (int*)alloc(512);
    int*       csr_src   = (int*)alloc((size_t)N_EDGES * sizeof(int)); // 3.2 MB
    _Float16*  Wh_src    = (_Float16*)alloc((size_t)IN_DIM * HD * sizeof(_Float16)); // 32 KB
    _Float16*  Wh_dst    = (_Float16*)alloc((size_t)IN_DIM * HD * sizeof(_Float16)); // 32 KB
    _Float16*  Wcomb     = (_Float16*)alloc((size_t)2 * 16 * IN_DIM * sizeof(_Float16)); // 8 KB
    float*     bcomb     = (float*)alloc((size_t)2 * 16 * sizeof(float));

    const int NB = (N_DST + SCAN_T - 1) / SCAN_T;                  // 98 (<=128)

    hipLaunchKernelGGL(setup_weights, dim3((IN_DIM * HD + 255) / 256), dim3(256), 0, stream,
                       W_src, W_dst, b_src, b_dst, attn, Wh_src, Wh_dst, Wcomb, bcomb);
    dim3 gProj((N_SRC / 16 + 7) / 8, 2);
    hipLaunchKernelGGL(gat_proj_kernel, gProj, dim3(256), 0, stream,
                       feat_src, feat_dst, Wh_src, b_src, Wh_dst, b_dst, Wcomb, bcomb,
                       h_src_buf, el_buf, er_buf);
    hipLaunchKernelGGL(zero_i32, dim3((N_DST + 255) / 256), dim3(256), 0, stream, counts, N_DST);
    hipLaunchKernelGGL(edge_count, dim3((N_EDGES + 255) / 256), dim3(256), 0, stream, dst_idx, counts);
    hipLaunchKernelGGL(scan1, dim3(NB), dim3(SCAN_T), 0, stream, counts, offs, bsums);
    hipLaunchKernelGGL(scan2, dim3(1), dim3(128), 0, stream, bsums, NB);
    hipLaunchKernelGGL(scan3, dim3(NB), dim3(SCAN_T), 0, stream, offs, bsums, counts);
    hipLaunchKernelGGL(edge_fill, dim3((N_EDGES + 255) / 256), dim3(256), 0, stream,
                       src_idx, dst_idx, counts, csr_src);
    hipLaunchKernelGGL(gat_agg, dim3((N_DST + 7) / 8), dim3(256), 0, stream,
                       h_src_buf, el_buf, er_buf, offs, csr_src, out);
}
